// VanillaTransformer_59717225283659
// MI455X (gfx1250) — matrix operand order repaired and checked by reading, not compiled, not run on hardware
//
#include <hip/hip_runtime.h>
#include <math.h>

typedef __attribute__((ext_vector_type(16))) _Float16 v16h;
typedef __attribute__((ext_vector_type(8)))  _Float16 v8h;
typedef __attribute__((ext_vector_type(4)))  _Float16 v4h;
typedef __attribute__((ext_vector_type(8)))  float    v8f;

#define D_MODEL 512
#define NHEAD   8
#define NLAYER  8
#define DFF_    2048
#define TVOCAB  32001
#define BATCH   8
#define SRCLEN  512
#define TGTLEN  256
#define LN_EPS  1e-6f

// ---------------------------------------------------------------------------
// WMMA fragment helpers (CDNA5 gfx1250, wave32). See ISA 7.12.2 layouts.
// ---------------------------------------------------------------------------
__device__ __forceinline__ v16h load_frag(const _Float16* p0, const _Float16* p1) {
  v16h r;
  v8h lo = *(const v8h*)p0;
  v8h hi = *(const v8h*)p1;
#pragma unroll
  for (int i = 0; i < 8; ++i) { r[i] = lo[i]; r[i + 8] = hi[i]; }
  return r;
}

__device__ __forceinline__ v8f wmma16(v16h a, v16h b, v8f c) {
  return __builtin_amdgcn_wmma_f32_16x16x32_f16(false, a, false, b, (short)0, c,
                                                false, false);
}

__device__ __forceinline__ void st4h(_Float16* p, float x, float y, float z, float w) {
  v4h t;
  t[0] = (_Float16)x; t[1] = (_Float16)y; t[2] = (_Float16)z; t[3] = (_Float16)w;
  *(v4h*)p = t;
}

// ---------------------------------------------------------------------------
// GEMM: C[M,N] = act(A[M,K] @ W[K,N] + bias[N]) (+ residual[M,N])
// Block = 256 threads (8 waves); tile 64(M) x 128(N), K-step 32.
// Each wave owns a 32x32 sub-tile (2x2 WMMA tiles).
// GUARDED=false: N%4==0 and N%TN==0 -> unconditional b128 W loads.
// GUARDED=true (generator, N=32001): branch-free clamped loads + select.
// ---------------------------------------------------------------------------
#define TM 64
#define TN 128
#define TK 32
#define LDP 40   // padded LDS row stride (halves): 16B-aligned frags, fewer bank conflicts

template <bool GUARDED>
__global__ __launch_bounds__(256) void gemm_wmma(
    const float* __restrict__ A, const float* __restrict__ W,
    const float* __restrict__ bias, const float* __restrict__ resid,
    float* __restrict__ C, int M, int N, int K, int relu) {
  __shared__ __align__(16) _Float16 As[TM][LDP];   // [m][k]
  __shared__ __align__(16) _Float16 Bs[TN][LDP];   // transposed: [n][k]

  const int tid  = threadIdx.x;
  const int lane = tid & 31, wave = tid >> 5;
  const int row16 = lane & 15, half = lane >> 4;
  const int wr = (wave >> 2) * 32, wc = (wave & 3) * 32;
  const int rowBase = blockIdx.y * TM;
  const int colBase = blockIdx.x * TN;

  v8f acc[2][2] = {};

  for (int k0 = 0; k0 < K; k0 += TK) {
    // ---- phase 1: gather to registers (independent, batched loads) ----
    float4 aR[2];
    int ar[2], akc[2];
#pragma unroll
    for (int i = 0; i < 2; ++i) {            // A tile: 64x32 = 512 float4
      int j = tid + i * 256;
      ar[i] = j >> 3; akc[i] = (j & 7) * 4;
      aR[i] = *(const float4*)&A[(size_t)(rowBase + ar[i]) * K + (k0 + akc[i])];
    }
    float4 bR[4];
    int bkk[4], bn0[4];
#pragma unroll
    for (int i = 0; i < 4; ++i) {            // W tile: 32x128 = 1024 float4
      int j = tid + i * 256;
      bkk[i] = j >> 5; bn0[i] = (j & 31) * 4;
      int gc = colBase + bn0[i];
      const float* wrow = &W[(size_t)(k0 + bkk[i]) * N];
      if (!GUARDED) {
        bR[i] = *(const float4*)&wrow[gc];
      } else {
        // clamped addresses (always in-bounds) + branch-free zero-select
        int m1 = N - 1;
        float x = wrow[(gc + 0 < N) ? gc + 0 : m1];
        float y = wrow[(gc + 1 < N) ? gc + 1 : m1];
        float z = wrow[(gc + 2 < N) ? gc + 2 : m1];
        float w = wrow[(gc + 3 < N) ? gc + 3 : m1];
        bR[i].x = (gc + 0 < N) ? x : 0.0f;
        bR[i].y = (gc + 1 < N) ? y : 0.0f;
        bR[i].z = (gc + 2 < N) ? z : 0.0f;
        bR[i].w = (gc + 3 < N) ? w : 0.0f;
      }
    }
    __syncthreads();   // previous iteration's fragment loads complete

    // ---- phase 2: convert + store to LDS ----
#pragma unroll
    for (int i = 0; i < 2; ++i)
      st4h(&As[ar[i]][akc[i]], aR[i].x, aR[i].y, aR[i].z, aR[i].w);
#pragma unroll
    for (int i = 0; i < 4; ++i) {            // transposed store
      Bs[bn0[i] + 0][bkk[i]] = (_Float16)bR[i].x;
      Bs[bn0[i] + 1][bkk[i]] = (_Float16)bR[i].y;
      Bs[bn0[i] + 2][bkk[i]] = (_Float16)bR[i].z;
      Bs[bn0[i] + 3][bkk[i]] = (_Float16)bR[i].w;
    }
    __syncthreads();

    // ---- phase 3: fragments + WMMA ----
    v16h af0 = load_frag(&As[wr + row16][half * 8],      &As[wr + row16][16 + half * 8]);
    v16h af1 = load_frag(&As[wr + 16 + row16][half * 8], &As[wr + 16 + row16][16 + half * 8]);
    v16h bf0 = load_frag(&Bs[wc + row16][half * 8],      &Bs[wc + row16][16 + half * 8]);
    v16h bf1 = load_frag(&Bs[wc + 16 + row16][half * 8], &Bs[wc + 16 + row16][16 + half * 8]);

    acc[0][0] = wmma16(af0, bf0, acc[0][0]);
    acc[0][1] = wmma16(af0, bf1, acc[0][1]);
    acc[1][0] = wmma16(af1, bf0, acc[1][0]);
    acc[1][1] = wmma16(af1, bf1, acc[1][1]);
  }

#pragma unroll
  for (int mt = 0; mt < 2; ++mt)
#pragma unroll
    for (int nt = 0; nt < 2; ++nt)
#pragma unroll
      for (int r = 0; r < 8; ++r) {
        int gr = rowBase + wr + mt * 16 + r + half * 8;
        int gc = colBase + wc + nt * 16 + row16;
        if (gr < M && gc < N) {
          float v = acc[mt][nt][r] + bias[gc];
          if (relu) v = fmaxf(v, 0.0f);
          if (resid) v += resid[(size_t)gr * N + gc];
          C[(size_t)gr * N + gc] = v;
        }
      }
}

// ---------------------------------------------------------------------------
// Fused attention: Out = softmax(Q Kᵀ/√64, key mask) V
// grid = (Sq/64, H, B), block = 128 (4 waves, 16 q-rows per wave).
// Flash-style online softmax over 32-key tiles; QKᵀ and PV via WMMA.
// ---------------------------------------------------------------------------
#define QP 72   // Qs/Ks padded row stride (halves)
#define VP 40   // VsT padded row stride

__global__ __launch_bounds__(128) void attn_wmma(
    const float* __restrict__ Q, const float* __restrict__ Km,
    const float* __restrict__ Vm, const int* __restrict__ mask,
    float* __restrict__ Out, int Sq, int Sk) {
  const int b = blockIdx.z, h = blockIdx.y, q0 = blockIdx.x * 64;
  __shared__ __align__(16) _Float16 Qs[64][QP];     // [q][d], pre-scaled
  __shared__ __align__(16) _Float16 Ks[32][QP];     // [key][d]
  __shared__ __align__(16) _Float16 VsT[64][VP];    // [d][key]
  __shared__ __align__(16) _Float16 Pw[4][16][32];  // per-wave P staging

  const int tid  = threadIdx.x;
  const int lane = tid & 31, wave = tid >> 5;
  const int row16 = lane & 15, half = lane >> 4;

#pragma unroll
  for (int i = 0; i < 8; ++i) {                     // Q tile: 64x64 = 1024 float4
    int j = tid + i * 128;
    int r = j >> 4, d0 = (j & 15) * 4;
    float4 qv = *(const float4*)&Q[(size_t)(b * Sq + q0 + r) * D_MODEL + h * 64 + d0];
    st4h(&Qs[r][d0], qv.x * 0.125f, qv.y * 0.125f, qv.z * 0.125f, qv.w * 0.125f);
  }

  float mrow[8], lrow[8];
  v8f acc[4] = {};
#pragma unroll
  for (int r = 0; r < 8; ++r) { mrow[r] = -1e30f; lrow[r] = 0.0f; }

  for (int kt = 0; kt < Sk; kt += 32) {
    // ---- gather K/V tiles to registers ----
    float4 kR[4], vR[4];
    int kky[4], kd0[4];
#pragma unroll
    for (int i = 0; i < 4; ++i) {                   // 32x64 = 512 float4 each
      int j = tid + i * 128;
      kky[i] = j >> 4; kd0[i] = (j & 15) * 4;
      size_t base = (size_t)(b * Sk + kt + kky[i]) * D_MODEL + h * 64 + kd0[i];
      kR[i] = *(const float4*)&Km[base];
      vR[i] = *(const float4*)&Vm[base];
    }
    __syncthreads();   // previous iteration's LDS consumers done
#pragma unroll
    for (int i = 0; i < 4; ++i) {
      st4h(&Ks[kky[i]][kd0[i]], kR[i].x, kR[i].y, kR[i].z, kR[i].w);
      VsT[kd0[i] + 0][kky[i]] = (_Float16)vR[i].x;  // transposed store
      VsT[kd0[i] + 1][kky[i]] = (_Float16)vR[i].y;
      VsT[kd0[i] + 2][kky[i]] = (_Float16)vR[i].z;
      VsT[kd0[i] + 3][kky[i]] = (_Float16)vR[i].w;
    }
    __syncthreads();

    // ---- scores: S = Q Kᵀ (K-dim 64 = 2 x 32) ----
    v8f s0 = {}, s1 = {};
#pragma unroll
    for (int ks = 0; ks < 2; ++ks) {
      v16h aq = load_frag(&Qs[wave * 16 + row16][ks * 32 + half * 8],
                          &Qs[wave * 16 + row16][ks * 32 + 16 + half * 8]);
      v16h b0 = load_frag(&Ks[row16][ks * 32 + half * 8],
                          &Ks[row16][ks * 32 + 16 + half * 8]);
      v16h b1 = load_frag(&Ks[16 + row16][ks * 32 + half * 8],
                          &Ks[16 + row16][ks * 32 + 16 + half * 8]);
      s0 = wmma16(aq, b0, s0);
      s1 = wmma16(aq, b1, s1);
    }

    int mk0 = mask[b * Sk + kt + row16];
    int mk1 = mask[b * Sk + kt + 16 + row16];
    if (!mk0) {
#pragma unroll
      for (int r = 0; r < 8; ++r) s0[r] = -1e9f;
    }
    if (!mk1) {
#pragma unroll
      for (int r = 0; r < 8; ++r) s1[r] = -1e9f;
    }

    // ---- online softmax (row stats via 16-lane-half butterflies) ----
#pragma unroll
    for (int r = 0; r < 8; ++r) {
      float mx = fmaxf(s0[r], s1[r]);
      mx = fmaxf(mx, __shfl_xor(mx, 1, 32));
      mx = fmaxf(mx, __shfl_xor(mx, 2, 32));
      mx = fmaxf(mx, __shfl_xor(mx, 4, 32));
      mx = fmaxf(mx, __shfl_xor(mx, 8, 32));
      float mnew = fmaxf(mrow[r], mx);
      float sc = __expf(mrow[r] - mnew);
      float p0 = __expf(s0[r] - mnew);
      float p1 = __expf(s1[r] - mnew);
      float rs = p0 + p1;
      rs += __shfl_xor(rs, 1, 32);
      rs += __shfl_xor(rs, 2, 32);
      rs += __shfl_xor(rs, 4, 32);
      rs += __shfl_xor(rs, 8, 32);
      lrow[r] = lrow[r] * sc + rs;
      mrow[r] = mnew;
#pragma unroll
      for (int nt = 0; nt < 4; ++nt) acc[nt][r] *= sc;
      int prow = r + half * 8;
      Pw[wave][prow][row16]      = (_Float16)p0;   // accum layout -> LDS
      Pw[wave][prow][16 + row16] = (_Float16)p1;
    }
    asm volatile("s_wait_dscnt 0" ::: "memory");   // per-wave LDS RAW

    // ---- O += P V ----
    v16h ap = load_frag(&Pw[wave][row16][half * 8],
                        &Pw[wave][row16][16 + half * 8]);
#pragma unroll
    for (int nt = 0; nt < 4; ++nt) {
      v16h bv = load_frag(&VsT[nt * 16 + row16][half * 8],
                          &VsT[nt * 16 + row16][16 + half * 8]);
      acc[nt] = wmma16(ap, bv, acc[nt]);
    }
  }

#pragma unroll
  for (int nt = 0; nt < 4; ++nt)
#pragma unroll
    for (int r = 0; r < 8; ++r) {
      int qg = q0 + wave * 16 + r + half * 8;
      int d  = nt * 16 + row16;
      Out[(size_t)(b * Sq + qg) * D_MODEL + h * 64 + d] = acc[nt][r] / lrow[r];
    }
}

// ---------------------------------------------------------------------------
// LayerNorm (torch variant: unbiased std, divide by std+eps). Wave per row.
// Each lane owns 16 contiguous floats -> 4 float4 loads.
// ---------------------------------------------------------------------------
__global__ __launch_bounds__(256) void layernorm_k(
    const float* __restrict__ X, const float* __restrict__ g,
    const float* __restrict__ bta, float* __restrict__ Y, int rows) {
  int wave = threadIdx.x >> 5, lane = threadIdx.x & 31;
  int row = blockIdx.x * 8 + wave;
  if (row >= rows) return;
  const float* x = X + (size_t)row * D_MODEL + lane * 16;
  float4 v[4];
  float s = 0.0f;
#pragma unroll
  for (int i = 0; i < 4; ++i) {
    v[i] = *(const float4*)&x[i * 4];
    s += v[i].x + v[i].y + v[i].z + v[i].w;
  }
  s += __shfl_xor(s, 1, 32); s += __shfl_xor(s, 2, 32);
  s += __shfl_xor(s, 4, 32); s += __shfl_xor(s, 8, 32);
  s += __shfl_xor(s, 16, 32);
  float mean = s * (1.0f / D_MODEL);
  float ss = 0.0f;
#pragma unroll
  for (int i = 0; i < 4; ++i) {
    float a = v[i].x - mean, b2 = v[i].y - mean, c = v[i].z - mean, d = v[i].w - mean;
    ss += a * a + b2 * b2 + c * c + d * d;
  }
  ss += __shfl_xor(ss, 1, 32); ss += __shfl_xor(ss, 2, 32);
  ss += __shfl_xor(ss, 4, 32); ss += __shfl_xor(ss, 8, 32);
  ss += __shfl_xor(ss, 16, 32);
  float inv = 1.0f / (sqrtf(ss / (D_MODEL - 1)) + LN_EPS);
  float* y = Y + (size_t)row * D_MODEL + lane * 16;
  const float* gp = g + lane * 16;
  const float* bp = bta + lane * 16;
#pragma unroll
  for (int i = 0; i < 4; ++i) {
    float4 go = *(const float4*)&gp[i * 4];
    float4 bo = *(const float4*)&bp[i * 4];
    float4 o;
    o.x = go.x * (v[i].x - mean) * inv + bo.x;
    o.y = go.y * (v[i].y - mean) * inv + bo.y;
    o.z = go.z * (v[i].z - mean) * inv + bo.z;
    o.w = go.w * (v[i].w - mean) * inv + bo.w;
    *(float4*)&y[i * 4] = o;
  }
}

// ---------------------------------------------------------------------------
// Embedding * sqrt(D) + positional encoding. One block (128 thr) per token.
// ---------------------------------------------------------------------------
__global__ __launch_bounds__(128) void embed_k(
    const int* __restrict__ tok, const float* __restrict__ emb,
    const float* __restrict__ pe, float* __restrict__ out, int S) {
  int t = blockIdx.x;
  int pos = t % S;
  int id = tok[t];
  int d0 = threadIdx.x * 4;
  float4 e = *(const float4*)&emb[(size_t)id * D_MODEL + d0];
  float4 p = *(const float4*)&pe[(size_t)pos * D_MODEL + d0];
  float4 o;
  o.x = e.x * 22.62741699796952f + p.x;
  o.y = e.y * 22.62741699796952f + p.y;
  o.z = e.z * 22.62741699796952f + p.z;
  o.w = e.w * 22.62741699796952f + p.w;
  *(float4*)&out[(size_t)t * D_MODEL + d0] = o;
}

// ---------------------------------------------------------------------------
// In-place log_softmax over rows of length N. One block per row.
// ---------------------------------------------------------------------------
__global__ __launch_bounds__(256) void logsoftmax_k(float* __restrict__ X, int N) {
  __shared__ float red[256];
  float* x = X + (size_t)blockIdx.x * N;
  float mx = -1e30f;
  for (int i = threadIdx.x; i < N; i += 256) mx = fmaxf(mx, x[i]);
  red[threadIdx.x] = mx;
  __syncthreads();
  for (int s = 128; s > 0; s >>= 1) {
    if (threadIdx.x < s) red[threadIdx.x] = fmaxf(red[threadIdx.x], red[threadIdx.x + s]);
    __syncthreads();
  }
  mx = red[0];
  __syncthreads();
  float sum = 0.0f;
  for (int i = threadIdx.x; i < N; i += 256) sum += __expf(x[i] - mx);
  red[threadIdx.x] = sum;
  __syncthreads();
  for (int s = 128; s > 0; s >>= 1) {
    if (threadIdx.x < s) red[threadIdx.x] += red[threadIdx.x + s];
    __syncthreads();
  }
  float lse = mx + logf(red[0]);
  for (int i = threadIdx.x; i < N; i += 256) x[i] -= lse;
}

// ---------------------------------------------------------------------------
extern "C" void kernel_launch(void* const* d_in, const int* in_sizes, int n_in,
                              void* d_out, int out_size, void* d_ws, size_t ws_size,
                              hipStream_t stream) {
  (void)in_sizes; (void)n_in; (void)out_size; (void)ws_size;
  const int*   src       = (const int*)d_in[0];
  const int*   tgt       = (const int*)d_in[1];
  const int*   src_mask  = (const int*)d_in[2];
  // d_in[3] = tgt_mask (unused; decoder self-attn skipped in source)
  const float* emb_src   = (const float*)d_in[4];
  const float* emb_tgt   = (const float*)d_in[5];
  const float* pe        = (const float*)d_in[6];
  const float* e_qkvo_w  = (const float*)d_in[7];
  const float* e_qkvo_b  = (const float*)d_in[8];
  const float* e_ff_w1   = (const float*)d_in[9];
  const float* e_ff_b1   = (const float*)d_in[10];
  const float* e_ff_w2   = (const float*)d_in[11];
  const float* e_ff_b2   = (const float*)d_in[12];
  const float* e_ln_g    = (const float*)d_in[13];
  const float* e_ln_b    = (const float*)d_in[14];
  const float* e_norm_g  = (const float*)d_in[15];
  const float* e_norm_b  = (const float*)d_in[16];
  const float* d_qkvo_w  = (const float*)d_in[17];
  const float* d_qkvo_b  = (const float*)d_in[18];
  const float* d_ff_w1   = (const float*)d_in[19];
  const float* d_ff_b1   = (const float*)d_in[20];
  const float* d_ff_w2   = (const float*)d_in[21];
  const float* d_ff_b2   = (const float*)d_in[22];
  const float* d_ln_g    = (const float*)d_in[23];
  const float* d_ln_b    = (const float*)d_in[24];
  const float* d_norm_g  = (const float*)d_in[25];
  const float* d_norm_b  = (const float*)d_in[26];
  const float* gen_w     = (const float*)d_in[27];
  const float* gen_b     = (const float*)d_in[28];
  float* out = (float*)d_out;

  const size_t ME = (size_t)BATCH * SRCLEN;   // 4096 encoder tokens
  const size_t MD = (size_t)BATCH * TGTLEN;   // 2048 decoder tokens

  // workspace layout (fp32)
  float* ws   = (float*)d_ws;
  float* xe   = ws;                 ws += ME * D_MODEL;  // encoder residual stream
  float* he   = ws;                 ws += ME * D_MODEL;  // LN output
  float* qb   = ws;                 ws += ME * D_MODEL;
  float* kb   = ws;                 ws += ME * D_MODEL;
  float* vb   = ws;                 ws += ME * D_MODEL;
  float* att  = ws;                 ws += ME * D_MODEL;
  float* mem  = ws;                 ws += ME * D_MODEL;  // encoder memory
  float* ffh  = ws;                 ws += ME * DFF_;     // FFN hidden (shared enc/dec)
  float* yd   = ws;                 ws += MD * D_MODEL;  // decoder residual stream
  float* hd   = ws;                 ws += MD * D_MODEL;
  float* qd   = ws;                 ws += MD * D_MODEL;
  float* attd = ws;                 ws += MD * D_MODEL;

  auto gemm = [&](const float* A, const float* W, const float* bias,
                  const float* res, float* C, int M, int N, int K, int relu) {
    dim3 grid((N + TN - 1) / TN, M / TM);
    if ((N & 3) == 0 && (N % TN) == 0)
      gemm_wmma<false><<<grid, 256, 0, stream>>>(A, W, bias, res, C, M, N, K, relu);
    else
      gemm_wmma<true><<<grid, 256, 0, stream>>>(A, W, bias, res, C, M, N, K, relu);
  };

  // ---------------- encoder ----------------
  embed_k<<<(int)ME, 128, 0, stream>>>(src, emb_src, pe, xe, SRCLEN);
  for (int l = 0; l < NLAYER; ++l) {
    const float* W  = e_qkvo_w + (size_t)l * 4 * D_MODEL * D_MODEL;
    const float* Wb = e_qkvo_b + (size_t)l * 4 * D_MODEL;
    layernorm_k<<<(int)(ME / 8), 256, 0, stream>>>(
        xe, e_ln_g + (size_t)(l * 2 + 0) * D_MODEL, e_ln_b + (size_t)(l * 2 + 0) * D_MODEL, he, (int)ME);
    gemm(he, W + 0 * D_MODEL * D_MODEL, Wb + 0 * D_MODEL, nullptr, qb, (int)ME, D_MODEL, D_MODEL, 0);
    gemm(he, W + 1 * D_MODEL * D_MODEL, Wb + 1 * D_MODEL, nullptr, kb, (int)ME, D_MODEL, D_MODEL, 0);
    gemm(he, W + 2 * D_MODEL * D_MODEL, Wb + 2 * D_MODEL, nullptr, vb, (int)ME, D_MODEL, D_MODEL, 0);
    attn_wmma<<<dim3(SRCLEN / 64, NHEAD, BATCH), 128, 0, stream>>>(
        qb, kb, vb, src_mask, att, SRCLEN, SRCLEN);
    gemm(att, W + 3 * D_MODEL * D_MODEL, Wb + 3 * D_MODEL, xe, xe, (int)ME, D_MODEL, D_MODEL, 0);
    layernorm_k<<<(int)(ME / 8), 256, 0, stream>>>(
        xe, e_ln_g + (size_t)(l * 2 + 1) * D_MODEL, e_ln_b + (size_t)(l * 2 + 1) * D_MODEL, he, (int)ME);
    gemm(he, e_ff_w1 + (size_t)l * D_MODEL * DFF_, e_ff_b1 + (size_t)l * DFF_, nullptr, ffh,
         (int)ME, DFF_, D_MODEL, 1);
    gemm(ffh, e_ff_w2 + (size_t)l * DFF_ * D_MODEL, e_ff_b2 + (size_t)l * D_MODEL, xe, xe,
         (int)ME, D_MODEL, DFF_, 0);
  }
  layernorm_k<<<(int)(ME / 8), 256, 0, stream>>>(xe, e_norm_g, e_norm_b, mem, (int)ME);

  // ---------------- decoder (cross-attn + FFN only, per source) ----------------
  embed_k<<<(int)MD, 128, 0, stream>>>(tgt, emb_tgt, pe, yd, TGTLEN);
  for (int l = 0; l < NLAYER; ++l) {
    const float* W  = d_qkvo_w + (size_t)l * 4 * D_MODEL * D_MODEL;
    const float* Wb = d_qkvo_b + (size_t)l * 4 * D_MODEL;
    layernorm_k<<<(int)(MD / 8), 256, 0, stream>>>(
        yd, d_ln_g + (size_t)(l * 2 + 0) * D_MODEL, d_ln_b + (size_t)(l * 2 + 0) * D_MODEL, hd, (int)MD);
    gemm(hd,  W + 0 * D_MODEL * D_MODEL, Wb + 0 * D_MODEL, nullptr, qd, (int)MD, D_MODEL, D_MODEL, 0);
    gemm(mem, W + 1 * D_MODEL * D_MODEL, Wb + 1 * D_MODEL, nullptr, kb, (int)ME, D_MODEL, D_MODEL, 0);
    gemm(mem, W + 2 * D_MODEL * D_MODEL, Wb + 2 * D_MODEL, nullptr, vb, (int)ME, D_MODEL, D_MODEL, 0);
    attn_wmma<<<dim3(TGTLEN / 64, NHEAD, BATCH), 128, 0, stream>>>(
        qd, kb, vb, src_mask, attd, TGTLEN, SRCLEN);
    gemm(attd, W + 3 * D_MODEL * D_MODEL, Wb + 3 * D_MODEL, yd, yd, (int)MD, D_MODEL, D_MODEL, 0);
    layernorm_k<<<(int)(MD / 8), 256, 0, stream>>>(
        yd, d_ln_g + (size_t)(l * 2 + 1) * D_MODEL, d_ln_b + (size_t)(l * 2 + 1) * D_MODEL, hd, (int)MD);
    gemm(hd, d_ff_w1 + (size_t)l * D_MODEL * DFF_, d_ff_b1 + (size_t)l * DFF_, nullptr, ffh,
         (int)MD, DFF_, D_MODEL, 1);
    gemm(ffh, d_ff_w2 + (size_t)l * DFF_ * D_MODEL, d_ff_b2 + (size_t)l * D_MODEL, yd, yd,
         (int)MD, D_MODEL, DFF_, 0);
  }
  layernorm_k<<<(int)(MD / 8), 256, 0, stream>>>(yd, d_norm_g, d_norm_b, hd, (int)MD);

  // ---------------- generator ----------------
  gemm(hd, gen_w, gen_b, nullptr, out, (int)MD, TVOCAB, D_MODEL, 0);
  logsoftmax_k<<<(int)MD, 256, 0, stream>>>(out, TVOCAB);
}
